// EnsembleNet3_88965952569541
// MI455X (gfx1250) — compile-verified, matrix operand order checked by reading, not run on hardware
//
#include <hip/hip_runtime.h>

// ---------------- problem constants ----------------
#define BG   32          // batch (graphs)
#define NN   512         // nodes per graph
#define BN   16384       // BG*NN
#define WF   128         // hidden width
#define DIM2 1536
#define KG   100         // graph kNN
#define KP   3           // point kNN

typedef __attribute__((ext_vector_type(16))) _Float16 v16h;
typedef __attribute__((ext_vector_type(8)))  _Float16 v8h;
typedef __attribute__((ext_vector_type(8)))  float    v8f;

// =====================================================================
// Guard-free fp16 WMMA GEMM:  C[M,N] = act( A[M,K] * Bt[N,K]^T + bias )
//   - A row-major [M,K], Bt row-major [N,K]  (B pre-transposed)
//   - REQUIRES: K % 32 == 0, N % 16 == 0, lda/ldbt % 8 == 0, 16B-aligned
//     bases; for MT=4, M % 512 == 0 (block tile = 8 waves x 4 tiles x 16).
//   - Each wave owns MT m-tiles sharing one B fragment: 4 WMMAs per
//     B-fragment load; all operand fetches are 16-byte global_load_b128.
// Outputs: optional f32 C, optional f16 row-major mirror Ch, optional
// f16 *transposed* per-graph mirror ChT[(m>>9)*ntT + n][m&511] (feeds the
// next GEMM's Bt operand so dynamic B never needs strided loads).
// =====================================================================
template<int MT>
__global__ void gemm_wmma_f16(const _Float16* __restrict__ A, int lda, long long aBatch,
                              const _Float16* __restrict__ Bt, int ldbt, long long bBatch,
                              float* __restrict__ Cf, int ldc, long long cBatch,
                              _Float16* __restrict__ Ch, int ldch, long long chBatch,
                              _Float16* __restrict__ ChT, long long chTBatch, int ntT,
                              const float* __restrict__ bias,
                              int M, int N, int K, int lrelu)
{
    const int wave = threadIdx.x >> 5;
    const int lane = threadIdx.x & 31;
    const int m0 = blockIdx.y * (128 * MT) + wave * (16 * MT);
    const int n0 = blockIdx.x * 16;
    if (m0 >= M) return;                       // wave-uniform (only M=32, MT=1)

    const long long bo = blockIdx.z;
    const int lm   = lane & 15;
    const int koff = (lane < 16) ? 0 : 8;      // ISA 16-bit A/B lane K-offset

    const _Float16* Arow[MT];
#pragma unroll
    for (int t = 0; t < MT; ++t)
        Arow[t] = A + bo * aBatch + (long long)(m0 + 16 * t + lm) * lda;
    const _Float16* Brow = Bt + bo * bBatch + (long long)(n0 + lm) * ldbt;

    v8f acc[MT];
#pragma unroll
    for (int t = 0; t < MT; ++t) acc[t] = (v8f){};

    for (int kb = 0; kb < K; kb += 32) {
        v8h b0 = *(const v8h*)(Brow + kb + koff);
        v8h b1 = *(const v8h*)(Brow + kb + 16 + koff);
        v16h bf = __builtin_shufflevector(b0, b1, 0,1,2,3,4,5,6,7,8,9,10,11,12,13,14,15);
#pragma unroll
        for (int t = 0; t < MT; ++t) {
            v8h a0 = *(const v8h*)(Arow[t] + kb + koff);
            v8h a1 = *(const v8h*)(Arow[t] + kb + 16 + koff);
            v16h af = __builtin_shufflevector(a0, a1,
                                              0,1,2,3,4,5,6,7,8,9,10,11,12,13,14,15);
            acc[t] = __builtin_amdgcn_wmma_f32_16x16x32_f16(false, af, false, bf,
                                                            (short)0, acc[t],
                                                            false, false);
        }
    }

    const int col = n0 + lm;
#pragma unroll
    for (int t = 0; t < MT; ++t) {
        const int mbase = m0 + 16 * t + ((lane < 16) ? 0 : 8);  // D: VGPR r -> M=r/r+8
#pragma unroll
        for (int r = 0; r < 8; ++r) {
            int m = mbase + r;
            float v = acc[t][r];
            if (bias)  v += bias[col];
            if (lrelu) v = (v > 0.0f) ? v : 0.01f * v;
            if (Cf)  Cf[bo * cBatch  + (long long)m * ldc  + col] = v;
            if (Ch)  Ch[bo * chBatch + (long long)m * ldch + col] = (_Float16)v;
            if (ChT) ChT[bo * chTBatch +
                         ((long long)(m >> 9) * ntT + col) * NN + (m & (NN - 1))]
                     = (_Float16)v;
        }
    }
}

// ---------------- pairwise squared distances, small C (coords) ----------------
__global__ void dist_kernel(const float* __restrict__ X, float* __restrict__ D, int C)
{
    const int bi = blockIdx.x;            // global node index b*NN+i
    const int b  = bi >> 9;
    const int i  = bi & (NN - 1);
    __shared__ float xi[16];
    __shared__ float sqi;
    const float* xrow = X + (long long)bi * C;
    for (int c = threadIdx.x; c < C; c += blockDim.x) xi[c] = xrow[c];
    __syncthreads();
    if (threadIdx.x == 0) {
        float s = 0.f;
        for (int c = 0; c < C; ++c) s += xi[c] * xi[c];
        sqi = s;
    }
    __syncthreads();
    const float* Xb = X + (long long)b * NN * C;
    float* drow = D + (long long)bi * NN;
    for (int j = threadIdx.x; j < NN; j += blockDim.x) {
        const float* xj = Xb + (long long)j * C;
        float dot = 0.f, sq = 0.f;
        for (int c = 0; c < C; ++c) { float v = xj[c]; dot += xi[c] * v; sq += v * v; }
        float d = sqi + sq - 2.0f * dot;
        if (j == i) d += 1e10f;
        drow[j] = d;
    }
}

// Gram -> distance epilogue (large-C kNN path; Gram computed with WMMA)
__global__ void sqnorm_kernel(const float* __restrict__ Y, float* __restrict__ sq, int C)
{
    int t = blockIdx.x * blockDim.x + threadIdx.x;
    if (t >= BN) return;
    const float* p = Y + (long long)t * C;
    float s = 0.f;
    for (int c = 0; c < C; ++c) { float v = p[c]; s += v * v; }
    sq[t] = s;
}

__global__ void dist_epilogue(float* __restrict__ D, const float* __restrict__ sq)
{
    long long t = (long long)blockIdx.x * blockDim.x + threadIdx.x;
    if (t >= (long long)BG * NN * NN) return;
    long long b = t >> 18;                 // / (NN*NN)
    int i = (int)((t >> 9) & (NN - 1));
    int j = (int)(t & (NN - 1));
    float d = sq[b * NN + i] + sq[b * NN + j] - 2.0f * D[t];
    if (i == j) d += 1e10f;
    D[t] = d;
}

// ---------------- top-K smallest per row (iterative LDS argmin) ----------------
__global__ void topk_kernel(const float* __restrict__ D, int* __restrict__ idx, int K)
{
    const int bi = blockIdx.x;
    __shared__ float row[NN];
    __shared__ float sval[256];
    __shared__ int   sidx[256];
    const float* drow = D + (long long)bi * NN;
    for (int j = threadIdx.x; j < NN; j += 256) row[j] = drow[j];
    __syncthreads();
    int* orow = idx + (long long)bi * K;
    for (int it = 0; it < K; ++it) {
        float best = 3.0e38f; int bj = NN;
        for (int j = threadIdx.x; j < NN; j += 256) {
            float v = row[j];
            if (v < best || (v == best && j < bj)) { best = v; bj = j; }
        }
        sval[threadIdx.x] = best; sidx[threadIdx.x] = bj;
        __syncthreads();
        for (int s = 128; s > 0; s >>= 1) {
            if (threadIdx.x < s) {
                float v = sval[threadIdx.x + s]; int jx = sidx[threadIdx.x + s];
                if (v < sval[threadIdx.x] ||
                    (v == sval[threadIdx.x] && jx < sidx[threadIdx.x])) {
                    sval[threadIdx.x] = v; sidx[threadIdx.x] = jx;
                }
            }
            __syncthreads();
        }
        if (threadIdx.x == 0) { orow[it] = sidx[0]; row[sidx[0]] = 3.0e38f; }
        __syncthreads();
    }
}

// ---------------- misc small kernels ----------------
__global__ void zero_u32(unsigned int* p, long long n)
{
    long long i = (long long)blockIdx.x * blockDim.x + threadIdx.x;
    long long st = (long long)gridDim.x * blockDim.x;
    for (; i < n; i += st) p[i] = 0u;
}

__global__ void scatter_S(const int* __restrict__ idx100, _Float16* __restrict__ S)
{
    long long t = (long long)blockIdx.x * blockDim.x + threadIdx.x;
    if (t >= (long long)BN * KG) return;
    long long node = t / KG;
    int j = idx100[t];
    S[node * NN + j] = (_Float16)0.01f;    // 1/k normalized adjacency
}

// f32 -> f16 with independent row strides
__global__ void cast_strided(const float* __restrict__ src, _Float16* __restrict__ dst,
                             long long total, int C, int sstride, int dstride)
{
    long long t = (long long)blockIdx.x * blockDim.x + threadIdx.x;
    if (t >= total) return;
    long long r = t / C; int c = (int)(t % C);
    dst[r * (long long)dstride + c] = (_Float16)src[r * (long long)sstride + c];
}

// weight transpose with row padding/segmenting:
// dst[n*ldd + g*gseg + r] = src[(g*rpg + r)*N + n]   (dst pre-zeroed if padded)
__global__ void transpose_pad(const float* __restrict__ src, _Float16* __restrict__ dst,
                              int groups, int rpg, int gseg, int N, int ldd)
{
    long long t = (long long)blockIdx.x * blockDim.x + threadIdx.x;
    long long total = (long long)groups * rpg * N;
    if (t >= total) return;
    int n = (int)(t % N);
    long long gr = t / N;
    int g = (int)(gr / rpg);
    int r = (int)(gr % rpg);
    dst[(long long)n * ldd + g * gseg + r] = (_Float16)src[gr * N + n];
}

// coords -> transposed fp16 [BG][16][NN] (rows c>=6 zero)
__global__ void xT_kernel(const float* __restrict__ x, _Float16* __restrict__ xT)
{
    long long t = (long long)blockIdx.x * blockDim.x + threadIdx.x;
    if (t >= (long long)BG * 16 * NN) return;
    int i = (int)(t & (NN - 1));
    int c = (int)((t >> 9) & 15);
    long long b = t >> 13;
    float v = (c < 6) ? x[(b * NN + i) * 6 + c] : 0.0f;
    xT[t] = (_Float16)v;
}

// edge features [yi | yj - yi] -> fp16 row of width ldE (zero-padded tail)
__global__ void build_E(const float* __restrict__ Y, const int* __restrict__ idx3,
                        _Float16* __restrict__ E, int C, int ldE)
{
    long long t = (long long)blockIdx.x * blockDim.x + threadIdx.x;
    long long total = (long long)BN * KP * C;
    if (t >= total) return;
    long long rc = t / C; int c = (int)(t % C);
    long long node = rc / KP; int j = (int)(rc % KP);
    long long b = node >> 9;
    int nb = idx3[node * KP + j];
    float yi = Y[node * C + c];
    float yj = Y[(b * NN + nb) * (long long)C + c];
    _Float16* erow = E + rc * (long long)ldE;
    erow[c]     = (_Float16)yi;
    erow[C + c] = (_Float16)(yj - yi);
}

__global__ void rowmax3(const float* __restrict__ u, float* __restrict__ y,
                        _Float16* __restrict__ yh, int C)
{
    long long t = (long long)blockIdx.x * blockDim.x + threadIdx.x;
    if (t >= (long long)BN * C) return;
    long long node = t / C; int c = (int)(t % C);
    const float* p = u + (node * KP) * (long long)C + c;
    float m = p[0];
    m = fmaxf(m, p[C]);
    m = fmaxf(m, p[2 * C]);
    y[t]  = m;
    yh[t] = (_Float16)m;
}

// per-graph mean/max over N nodes, written straight into z columns
__global__ void pool_kernel(const float* __restrict__ H, float* __restrict__ Z,
                            int C, int mean_col, int max_col)
{
    int b = blockIdx.x; int f = threadIdx.x;
    if (f >= C) return;
    const float* p = H + ((long long)b * NN) * C + f;
    float s = 0.f, mx = -3.0e38f;
    for (int i = 0; i < NN; ++i) { float v = p[(long long)i * C]; s += v; mx = fmaxf(mx, v); }
    Z[b * DIM2 + mean_col + f] = s * (1.0f / NN);
    Z[b * DIM2 + max_col  + f] = mx;
}

__global__ void bn_kernel(float* __restrict__ Z, _Float16* __restrict__ Zh,
                          const float* __restrict__ gamma, const float* __restrict__ beta)
{
    int c = blockIdx.x * blockDim.x + threadIdx.x;
    if (c >= DIM2) return;
    float s = 0.f;
    for (int b = 0; b < BG; ++b) s += Z[b * DIM2 + c];
    float mu = s * (1.0f / BG);
    float v = 0.f;
    for (int b = 0; b < BG; ++b) { float d = Z[b * DIM2 + c] - mu; v += d * d; }
    v *= (1.0f / BG);
    float inv = rsqrtf(v + 1e-5f);
    float g = gamma[c], be = beta[c];
    for (int b = 0; b < BG; ++b) {
        float z = (Z[b * DIM2 + c] - mu) * inv * g + be;
        Z[b * DIM2 + c]  = z;
        Zh[b * DIM2 + c] = (_Float16)z;
    }
}

__global__ void out_kernel(const float* __restrict__ Z, const float* __restrict__ oW,
                           const float* __restrict__ ob, float* __restrict__ out)
{
    int b = blockIdx.x;
    __shared__ float red[256];
    float s = 0.f;
    for (int c = threadIdx.x; c < DIM2; c += 256) s += Z[b * DIM2 + c] * oW[c];
    red[threadIdx.x] = s;
    __syncthreads();
    for (int st = 128; st > 0; st >>= 1) {
        if (threadIdx.x < st) red[threadIdx.x] += red[threadIdx.x + st];
        __syncthreads();
    }
    if (threadIdx.x == 0) out[b] = red[0] + ob[0];
}

// =====================================================================
static inline char* wsTake(char*& p, size_t bytes)
{
    char* r = p;
    p += (bytes + 255) & ~((size_t)255);
    return r;
}

extern "C" void kernel_launch(void* const* d_in, const int* in_sizes, int n_in,
                              void* d_out, int out_size, void* d_ws, size_t ws_size,
                              hipStream_t stream)
{
    (void)in_sizes; (void)n_in; (void)out_size; (void)ws_size;
    const float* x      = (const float*)d_in[0];
    const float* tag1_W = (const float*)d_in[2];
    const float* tag1_b = (const float*)d_in[3];
    const float* tag_W  = (const float*)d_in[4];
    const float* tag_b  = (const float*)d_in[5];
    const float* p1_W1  = (const float*)d_in[6];
    const float* p1_b1  = (const float*)d_in[7];
    const float* p1_W2  = (const float*)d_in[8];
    const float* p1_b2  = (const float*)d_in[9];
    const float* pf_W   = (const float*)d_in[10];
    const float* pf_b   = (const float*)d_in[11];
    const float* bn_g   = (const float*)d_in[12];
    const float* bn_b   = (const float*)d_in[13];
    const float* lin_W  = (const float*)d_in[14];
    const float* lin_b  = (const float*)d_in[15];
    const float* out_W  = (const float*)d_in[16];
    const float* out_b  = (const float*)d_in[17];
    float* out = (float*)d_out;

    char* wp = (char*)d_ws;
    float*     D     = (float*)    wsTake(wp, (size_t)BG * NN * NN * 4);
    _Float16*  S     = (_Float16*) wsTake(wp, (size_t)BG * NN * NN * 2);
    int*       i100  = (int*)      wsTake(wp, (size_t)BN * KG * 4);
    int*       i3    = (int*)      wsTake(wp, (size_t)BN * KP * 4);
    _Float16*  X31   = (_Float16*) wsTake(wp, (size_t)BN * 96 * 2);
    _Float16*  X3a   = (_Float16*) wsTake(wp, (size_t)BN * 384 * 2);
    _Float16*  X3b   = (_Float16*) wsTake(wp, (size_t)BN * 384 * 2);
    _Float16*  xT    = (_Float16*) wsTake(wp, (size_t)BG * 16 * NN * 2);
    _Float16*  m1T16 = (_Float16*) wsTake(wp, (size_t)BG * 16 * NN * 2);
    _Float16*  hT    = (_Float16*) wsTake(wp, (size_t)BG * WF * NN * 2);
    _Float16*  m1T   = (_Float16*) wsTake(wp, (size_t)BG * WF * NN * 2);
    _Float16*  Wc1T  = (_Float16*) wsTake(wp, (size_t)WF * 96 * 2);
    _Float16*  WcT   = (_Float16*) wsTake(wp, (size_t)WF * 384 * 2);
    float*     h     = (float*)    wsTake(wp, (size_t)BN * WF * 4);
    _Float16*  E     = (_Float16*) wsTake(wp, (size_t)BN * KP * 256 * 2);
    _Float16*  t_h   = (_Float16*) wsTake(wp, (size_t)BN * KP * WF * 2);
    float*     u     = (float*)    wsTake(wp, (size_t)BN * KP * WF * 4);
    float*     y     = (float*)    wsTake(wp, (size_t)BN * WF * 4);
    _Float16*  y_h   = (_Float16*) wsTake(wp, (size_t)BN * WF * 2);
    float*     sq    = (float*)    wsTake(wp, (size_t)BN * 4);
    _Float16*  W1eT  = (_Float16*) wsTake(wp, (size_t)WF * 32 * 2);
    _Float16*  W128T = (_Float16*) wsTake(wp, (size_t)WF * WF * 2);
    _Float16*  WpfT  = (_Float16*) wsTake(wp, (size_t)2 * WF * 256 * 2);
    _Float16*  WlinT = (_Float16*) wsTake(wp, (size_t)5 * DIM2 * DIM2 * 2);
    float*     z     = (float*)    wsTake(wp, (size_t)BG * DIM2 * 4);
    _Float16*  z_h   = (_Float16*) wsTake(wp, (size_t)BG * DIM2 * 2);
    float*     z2    = (float*)    wsTake(wp, (size_t)BG * DIM2 * 4);
    _Float16*  z2_h  = (_Float16*) wsTake(wp, (size_t)BG * DIM2 * 2);

    #define GB(n) (int)(((long long)(n) + 255) / 256)

    // ---- one-time weight transposes (Bt operands) ----
    zero_u32<<<GB(WF * 96 / 2), 256, 0, stream>>>((unsigned int*)Wc1T, WF * 96 / 2);
    transpose_pad<<<GB(18 * WF), 256, 0, stream>>>(tag1_W, Wc1T, 3, 6, 32, WF, 96);
    zero_u32<<<GB(WF * 32 / 2), 256, 0, stream>>>((unsigned int*)W1eT, WF * 32 / 2);
    transpose_pad<<<GB(12 * WF), 256, 0, stream>>>(p1_W1, W1eT, 1, 12, 32, WF, 32);
    transpose_pad<<<GB(WF * WF), 256, 0, stream>>>(p1_W2, W128T, 1, WF, WF, WF, WF);
    for (int f = 0; f < 2; ++f)
        transpose_pad<<<GB(256 * WF), 256, 0, stream>>>(pf_W + (size_t)f * 256 * WF,
                                                        WpfT + (size_t)f * WF * 256,
                                                        1, 256, 256, WF, 256);
    for (int g = 0; g < 5; ++g)
        transpose_pad<<<GB((long long)DIM2 * DIM2), 256, 0, stream>>>(
            lin_W + (size_t)g * DIM2 * DIM2, WlinT + (size_t)g * DIM2 * DIM2,
            1, DIM2, DIM2, DIM2, DIM2);

    // ---- kNN graph on raw coordinates (C=6) + dense normalized adjacency ----
    dist_kernel<<<BN, 256, 0, stream>>>(x, D, 6);
    topk_kernel<<<BN, 256, 0, stream>>>(D, i100, KG);
    topk_kernel<<<BN, 256, 0, stream>>>(D, i3, KP);   // EdgeConv #1 shares this metric
    zero_u32<<<2048, 256, 0, stream>>>((unsigned int*)S, (long long)BG * NN * NN / 2);
    scatter_S<<<GB((long long)BN * KG), 256, 0, stream>>>(i100, S);

    // ---- TAG layer 1:  X31 = [x | Sx | S^2 x] (segment-32), K=96 ----
    zero_u32<<<GB((long long)BN * 96 / 2), 256, 0, stream>>>((unsigned int*)X31,
                                                             (long long)BN * 96 / 2);
    cast_strided<<<GB(BN * 6), 256, 0, stream>>>(x, X31, (long long)BN * 6, 6, 6, 96);
    xT_kernel<<<GB((long long)BG * 16 * NN), 256, 0, stream>>>(x, xT);
    gemm_wmma_f16<4><<<dim3(1, 1, BG), 256, 0, stream>>>(
        S, NN, (long long)NN * NN, xT, NN, 16LL * NN,
        nullptr, 0, 0, X31 + 32, 96, 96LL * NN,
        m1T16, 16LL * NN, 16, nullptr, NN, 16, NN, 0);
    gemm_wmma_f16<4><<<dim3(1, 1, BG), 256, 0, stream>>>(
        S, NN, (long long)NN * NN, m1T16, NN, 16LL * NN,
        nullptr, 0, 0, X31 + 64, 96, 96LL * NN,
        nullptr, 0, 0, nullptr, NN, 16, NN, 0);
    gemm_wmma_f16<4><<<dim3(WF / 16, BN / 512, 1), 256, 0, stream>>>(
        X31, 96, 0, Wc1T, 96, 0, h, WF, 0, X3a, 384, 0,
        hT, 0, WF, tag1_b, BN, WF, 96, 1);
    pool_kernel<<<BG, WF, 0, stream>>>(h, z, WF, 0, 128);

    // ---- TAG layers 2,3: X3 = [h | Sh | S^2 h], K=384 fused combine ----
    for (int f = 0; f < 2; ++f) {
        _Float16* Xc = (f == 0) ? X3a : X3b;
        _Float16* Xn = (f == 0) ? X3b : nullptr;     // last layer needs no fp16 out
        transpose_pad<<<GB(384 * WF), 256, 0, stream>>>(
            tag_W + (size_t)f * 3 * WF * WF, WcT, 1, 384, 384, WF, 384);
        gemm_wmma_f16<4><<<dim3(WF / 16, 1, BG), 256, 0, stream>>>(
            S, NN, (long long)NN * NN, hT, NN, (long long)WF * NN,
            nullptr, 0, 0, Xc + 128, 384, 384LL * NN,
            m1T, (long long)WF * NN, WF, nullptr, NN, WF, NN, 0);
        gemm_wmma_f16<4><<<dim3(WF / 16, 1, BG), 256, 0, stream>>>(
            S, NN, (long long)NN * NN, m1T, NN, (long long)WF * NN,
            nullptr, 0, 0, Xc + 256, 384, 384LL * NN,
            nullptr, 0, 0, nullptr, NN, WF, NN, 0);
        gemm_wmma_f16<4><<<dim3(WF / 16, BN / 512, 1), 256, 0, stream>>>(
            Xc, 384, 0, WcT, 384, 0, h, WF, 0,
            Xn, 384, 0, (f == 0) ? hT : nullptr, 0, WF,
            tag_b + f * WF, BN, WF, 384, 1);
        pool_kernel<<<BG, WF, 0, stream>>>(h, z, WF, (f + 1) * 256, (f + 1) * 256 + 128);
    }

    // ---- EdgeConv 1 (kNN on coordinates), K=32 padded / K=128 ----
    zero_u32<<<GB((long long)BN * KP * 32 / 2), 256, 0, stream>>>(
        (unsigned int*)E, (long long)BN * KP * 32 / 2);
    build_E<<<GB((long long)BN * KP * 6), 256, 0, stream>>>(x, i3, E, 6, 32);
    gemm_wmma_f16<4><<<dim3(WF / 16, BN * KP / 512, 1), 256, 0, stream>>>(
        E, 32, 0, W1eT, 32, 0, nullptr, 0, 0, t_h, WF, 0,
        nullptr, 0, 0, p1_b1, BN * KP, WF, 32, 1);
    gemm_wmma_f16<4><<<dim3(WF / 16, BN * KP / 512, 1), 256, 0, stream>>>(
        t_h, WF, 0, W128T, WF, 0, u, WF, 0, nullptr, 0, 0,
        nullptr, 0, 0, p1_b2, BN * KP, WF, WF, 1);
    rowmax3<<<GB((long long)BN * WF), 256, 0, stream>>>(u, y, y_h, WF);
    pool_kernel<<<BG, WF, 0, stream>>>(y, z, WF, 768, 1152);

    // ---- EdgeConv 2,3: kNN via WMMA Gram matrix (A = Bt = y_h), K=256 MLP ----
    for (int f = 0; f < 2; ++f) {
        gemm_wmma_f16<4><<<dim3(NN / 16, 1, BG), 256, 0, stream>>>(
            y_h, WF, (long long)NN * WF, y_h, WF, (long long)NN * WF,
            D, NN, (long long)NN * NN, nullptr, 0, 0,
            nullptr, 0, 0, nullptr, NN, NN, WF, 0);
        sqnorm_kernel<<<GB(BN), 256, 0, stream>>>(y, sq, WF);
        dist_epilogue<<<GB((long long)BG * NN * NN), 256, 0, stream>>>(D, sq);
        topk_kernel<<<BN, 256, 0, stream>>>(D, i3, KP);
        build_E<<<GB((long long)BN * KP * WF), 256, 0, stream>>>(y, i3, E, WF, 256);
        gemm_wmma_f16<4><<<dim3(WF / 16, BN * KP / 512, 1), 256, 0, stream>>>(
            E, 256, 0, WpfT + (size_t)f * WF * 256, 256, 0, u, WF, 0,
            nullptr, 0, 0, nullptr, 0, 0, pf_b + f * WF, BN * KP, WF, 256, 1);
        rowmax3<<<GB((long long)BN * WF), 256, 0, stream>>>(u, y, y_h, WF);
        pool_kernel<<<BG, WF, 0, stream>>>(y, z, WF, 768 + (f + 1) * WF,
                                           1152 + (f + 1) * WF);
    }

    // ---- head: BN -> 5x (1536x1536) WMMA -> dot ----
    bn_kernel<<<GB(DIM2), 256, 0, stream>>>(z, z_h, bn_g, bn_b);
    for (int g = 0; g < 5; ++g) {
        const _Float16* Ain = (g & 1) ? z2_h : z_h;
        float*     Co = (g & 1) ? z : z2;
        _Float16*  Cm = (g & 1) ? z_h : z2_h;
        gemm_wmma_f16<1><<<dim3(DIM2 / 16, 1, 1), 256, 0, stream>>>(
            Ain, DIM2, 0, WlinT + (size_t)g * DIM2 * DIM2, DIM2, 0,
            Co, DIM2, 0, Cm, DIM2, 0, nullptr, 0, 0,
            lin_b + g * DIM2, BG, DIM2, DIM2, 1);
    }
    out_kernel<<<BG, 256, 0, stream>>>(z2, out_W, out_b, out);

    #undef GB
}
